// MaskedAttn_16690242913074
// MI455X (gfx1250) — matrix operand order repaired and checked by reading, not compiled, not run on hardware
//
#include <hip/hip_runtime.h>

typedef __attribute__((ext_vector_type(16))) _Float16 v16h;
typedef __attribute__((ext_vector_type(8)))  _Float16 v8h;
typedef __attribute__((ext_vector_type(4)))  _Float16 v4h;
typedef __attribute__((ext_vector_type(8)))  float    v8f;
typedef __attribute__((ext_vector_type(4)))  float    v4f;
typedef __attribute__((ext_vector_type(4)))  unsigned int u32x4;
typedef __attribute__((ext_vector_type(4)))  int      i32x4;
typedef __attribute__((ext_vector_type(8)))  int      i32x8;

// ---------------------------------------------------------------------------
// Fragment loaders matching the CDNA5 WMMA VGPR layouts (CDNA5 ISA)
// ---------------------------------------------------------------------------

// A-matrix 16x32 f16, source row-major [rows, ld].
__device__ __forceinline__ v16h load_A16(const _Float16* __restrict__ p, int ld,
                                         int row0, int k0) {
  const int lane = threadIdx.x & 31;
  const _Float16* base = p + (size_t)(row0 + (lane & 15)) * ld + k0 + ((lane >> 4) << 3);
  v8h lo = *(const v8h*)(base);
  v8h hi = *(const v8h*)(base + 16);
  v16h a;
#pragma unroll
  for (int i = 0; i < 8; ++i) { a[i] = lo[i]; a[8 + i] = hi[i]; }
  return a;
}

// B-matrix 32x16 f16 where B[k][n] = W[n0+n][k0+k], W row-major [cols, ld].
__device__ __forceinline__ v16h load_B16(const _Float16* __restrict__ p, int ld,
                                         int k0, int n0) {
  const int lane = threadIdx.x & 31;
  const _Float16* base = p + (size_t)(n0 + (lane & 15)) * ld + k0 + ((lane >> 4) << 3);
  v8h lo = *(const v8h*)(base);
  v8h hi = *(const v8h*)(base + 16);
  v16h b;
#pragma unroll
  for (int i = 0; i < 8; ++i) { b[i] = lo[i]; b[8 + i] = hi[i]; }
  return b;
}

__device__ __forceinline__ v8f wmma_f16(v16h a, v16h b, v8f c) {
  return __builtin_amdgcn_wmma_f32_16x16x32_f16(false, a, false, b, (short)0, c,
                                                false, false);
}

// ---------------------------------------------------------------------------
// 2-D tile staging into LDS (tile0 = contiguous elements per row, tile1 = rows,
// row stride ld; tile0*tile1 must equal 2048 f16 for the fallback path).
// TDM path (CDNA5 ISA): wave 0 issues tensor_load_to_lds
// with a D# built in SGPRs; pipeline syncs with s_wait_tensorcnt + barrier.
// ---------------------------------------------------------------------------
#define QKV_HAVE_TDM __has_builtin(__builtin_amdgcn_tensor_load_to_lds)

__device__ __forceinline__ void stage_issue(const _Float16* gsrc, int ld,
                                            _Float16* lbuf, int tile0, int tile1) {
#if QKV_HAVE_TDM
  if ((threadIdx.x >> 5) == 0) {
    const unsigned long long ga = (unsigned long long)(size_t)gsrc;
    const unsigned lds_off = (unsigned)(size_t)lbuf;  // flat LDS low 32 bits
    u32x4 g0 = {};
    g0[0] = 1u;                                        // count=1 (valid D#)
    g0[1] = lds_off;                                   // lds_addr
    g0[2] = (unsigned)ga;                              // global_addr[31:0]
    g0[3] = (unsigned)((ga >> 32) & 0x01FFFFFFu) | (2u << 30);  // addr hi | type=2
    i32x8 g1 = {};
    g1[0] = (int)(1u << 16);                           // data_size = 2 bytes
    g1[1] = (int)((unsigned)tile0 << 16);              // tensor_dim0 lo16
    g1[2] = (int)((unsigned)tile1 << 16);              // tdim0 hi=0 | tensor_dim1
    g1[3] = (int)((unsigned)tile0 << 16);              // tdim1 hi=0 | tile_dim0
    g1[4] = (int)(unsigned)tile1;                      // tile_dim1, tile_dim2=0
    g1[5] = (int)(unsigned)ld;                         // tensor_dim0_stride lo
    g1[6] = 0;
    g1[7] = 0;
    i32x4 z4 = {};
#if defined(__clang_major__) && __clang_major__ >= 23
    i32x8 z8 = {};
    __builtin_amdgcn_tensor_load_to_lds(g0, g1, z4, z4, z8, 0);
#else
    __builtin_amdgcn_tensor_load_to_lds(g0, g1, z4, z4, 0);
#endif
  }
#else
  const int t = threadIdx.x;                           // 256 threads x 16B
  const int cpr = tile0 >> 3;                          // 16B chunks per row
  const int r = t / cpr, c = (t % cpr) << 3;
  (void)tile1;
  *(v8h*)(lbuf + r * tile0 + c) = *(const v8h*)(gsrc + (size_t)r * ld + c);
#endif
}

__device__ __forceinline__ void stage_wait() {
#if QKV_HAVE_TDM
  if ((threadIdx.x >> 5) == 0) __builtin_amdgcn_s_wait_tensorcnt(0);
#endif
  __syncthreads();
}

// Double-buffered 128x64 GEMM core: 8 waves, each computing a 16x64 strip.
// A fragments from global (per wave), B tile TDM-staged in LDS (per block).
__device__ __forceinline__ void gemm_core(const _Float16* __restrict__ A, int lda,
                                          const _Float16* __restrict__ W, int ldw,
                                          int m0, int n0, int ksteps,
                                          _Float16* lds0, _Float16* lds1,
                                          v8f acc[4]) {
  stage_issue(W + (size_t)n0 * ldw, ldw, lds0, 32, 64);
  stage_wait();
  for (int ks = 0; ks < ksteps; ++ks) {
    _Float16* cur = (ks & 1) ? lds1 : lds0;
    _Float16* nxt = (ks & 1) ? lds0 : lds1;
    if (ks + 1 < ksteps)
      stage_issue(W + (size_t)n0 * ldw + (ks + 1) * 32, ldw, nxt, 32, 64);
    const v16h a = load_A16(A, lda, m0, ks * 32);
    acc[0] = wmma_f16(a, load_B16(cur, 32, 0, 0), acc[0]);
    acc[1] = wmma_f16(a, load_B16(cur, 32, 0, 16), acc[1]);
    acc[2] = wmma_f16(a, load_B16(cur, 32, 0, 32), acc[2]);
    acc[3] = wmma_f16(a, load_B16(cur, 32, 0, 48), acc[3]);
    stage_wait();
  }
}

// ---------------------------------------------------------------------------
// K0: f32 -> f16 conversion (128-bit accesses)
// ---------------------------------------------------------------------------
__global__ void cvt_f32_f16(const float* __restrict__ in, _Float16* __restrict__ out,
                            int n4) {
  int i = blockIdx.x * blockDim.x + threadIdx.x;
  if (i < n4) {
    v4f x = *(const v4f*)(in + 4 * (size_t)i);
    v4h o;
#pragma unroll
    for (int j = 0; j < 4; ++j) o[j] = (_Float16)x[j];
    *(v4h*)(out + 4 * (size_t)i) = o;
  }
}

// ---------------------------------------------------------------------------
// K1: qkv GEMM  [8192,768] x [2304,768]^T + bias -> q(scaled)/k row-major, v^T
// ---------------------------------------------------------------------------
__global__ __launch_bounds__(256) void qkv_gemm(
    const _Float16* __restrict__ xh,   // [8192, 768]
    const _Float16* __restrict__ wh,   // [2304, 768]
    const float* __restrict__ bias,    // [2304]
    _Float16* __restrict__ qh,         // [B,H,N,D] = [8,12,1024,64]
    _Float16* __restrict__ kh,         // [B,H,N,D]
    _Float16* __restrict__ vT) {       // [B,H,D,N]
  __shared__ __align__(64) _Float16 bsh[2][64 * 32];
  const int lane = threadIdx.x & 31;
  const int wslot = threadIdx.x >> 5;
  const int NB = 2304 / 64;                      // 36 col blocks
  const int m0 = (blockIdx.x / NB) * 128 + wslot * 16;
  const int n0 = (blockIdx.x % NB) * 64;

  v8f acc[4] = {{}, {}, {}, {}};
  gemm_core(xh, 768, wh, 768, m0, n0, 24, &bsh[0][0], &bsh[1][0], acc);

  const int side = lane >> 4;
#pragma unroll
  for (int j = 0; j < 4; ++j) {
    const int col = n0 + j * 16 + (lane & 15);
    const int which = col / 768;                 // 0=q 1=k 2=v (uniform per wave)
    const int cr = col % 768;
    const int h = cr >> 6, d = cr & 63;
    const float bv = bias[col];
#pragma unroll
    for (int r = 0; r < 8; ++r) {
      const int row = m0 + r + 8 * side;         // b*1024 + n
      const int b = row >> 10, n = row & 1023;
      const float v = acc[j][r] + bv;
      if (which == 0)
        qh[((size_t)(b * 12 + h) << 10 | n) * 64 + d] = (_Float16)(v * 0.125f);
      else if (which == 1)
        kh[((size_t)(b * 12 + h) << 10 | n) * 64 + d] = (_Float16)v;
      else
        vT[((size_t)(b * 12 + h) * 64 + d) * 1024 + n] = (_Float16)v;
    }
  }
}

// ---------------------------------------------------------------------------
// K2: flash attention. Block = 8 waves sharing one (b,h); each wave owns a
// 16-row q block. K/V tiles for each 32-key step are TDM-staged in LDS once
// per block (8x traffic reduction) with a double-buffered pipeline.
// ---------------------------------------------------------------------------
__global__ __launch_bounds__(256) void attn_fwd(
    const _Float16* __restrict__ qh, const _Float16* __restrict__ kh,
    const _Float16* __restrict__ vT, const float* __restrict__ mask,
    const long long* __restrict__ y, _Float16* __restrict__ oh /* [8192,768] */) {
  __shared__ __align__(64) _Float16 kbuf[2][32 * 64];  // [key][d]
  __shared__ __align__(64) _Float16 vbuf[2][64 * 32];  // [d][key]
  __shared__ __align__(32) _Float16 pb[8][16 * 32];    // per-wave P tile
  const int lane = threadIdx.x & 31;
  const int wslot = threadIdx.x >> 5;
  const int bh = blockIdx.x >> 3;                 // b*12 + h (8 blocks per bh)
  const int h = bh % 12, b = bh / 12;
  const int qb = (blockIdx.x & 7) * 8 + wslot;    // q block 0..63
  const _Float16* qp = qh + (size_t)bh * 1024 * 64;
  const _Float16* kp = kh + (size_t)bh * 1024 * 64;
  const _Float16* vp = vT + (size_t)bh * 64 * 1024;
  const int q0 = qb * 16;

  const v16h aq0 = load_A16(qp, 64, q0, 0);
  const v16h aq1 = load_A16(qp, 64, q0, 32);

  float mi[8], li[8];
  v8f o0 = {}, o1 = {}, o2 = {}, o3 = {};
#pragma unroll
  for (int r = 0; r < 8; ++r) { mi[r] = -3.0e38f; li[r] = 0.f; }
  _Float16* pbuf = &pb[wslot][0];
  const int side = lane >> 4;

  // prologue: stage K/V tiles for keys 0..31
  stage_issue(kp, 64, &kbuf[0][0], 64, 32);
  stage_issue(vp, 1024, &vbuf[0][0], 32, 64);
  stage_wait();

  for (int it = 0; it < 32; ++it) {
    const int n0 = it * 32;
    const _Float16* kcur = &kbuf[it & 1][0];
    const _Float16* vcur = &vbuf[it & 1][0];
    if (it + 1 < 32) {
      stage_issue(kp + (size_t)(n0 + 32) * 64, 64, &kbuf[(it + 1) & 1][0], 64, 32);
      stage_issue(vp + (n0 + 32), 1024, &vbuf[(it + 1) & 1][0], 32, 64);
    }

    v8f s0 = {}, s1 = {};
    s0 = wmma_f16(aq0, load_B16(kcur, 64, 0, 0), s0);
    s0 = wmma_f16(aq1, load_B16(kcur, 64, 32, 0), s0);
    s1 = wmma_f16(aq0, load_B16(kcur, 64, 0, 16), s1);
    s1 = wmma_f16(aq1, load_B16(kcur, 64, 32, 16), s1);

#pragma unroll
    for (int r = 0; r < 8; ++r) {
      float mx = fmaxf(s0[r], s1[r]);
      mx = fmaxf(mx, __shfl_xor(mx, 1, 32));
      mx = fmaxf(mx, __shfl_xor(mx, 2, 32));
      mx = fmaxf(mx, __shfl_xor(mx, 4, 32));
      mx = fmaxf(mx, __shfl_xor(mx, 8, 32));
      const float mnew = fmaxf(mi[r], mx);
      const float scl = __expf(mi[r] - mnew);
      const float p0 = __expf(s0[r] - mnew);
      const float p1 = __expf(s1[r] - mnew);
      float rs = p0 + p1;
      rs += __shfl_xor(rs, 1, 32);
      rs += __shfl_xor(rs, 2, 32);
      rs += __shfl_xor(rs, 4, 32);
      rs += __shfl_xor(rs, 8, 32);
      li[r] = li[r] * scl + rs;
      mi[r] = mnew;
      o0[r] *= scl; o1[r] *= scl; o2[r] *= scl; o3[r] *= scl;
      const int row = r + 8 * side;
      pbuf[row * 32 + (lane & 15)] = (_Float16)p0;
      pbuf[row * 32 + 16 + (lane & 15)] = (_Float16)p1;
    }
    asm volatile("s_wait_dscnt 0" ::: "memory");   // intra-wave LDS RAW fence
    const v16h ap = load_A16(pbuf, 32, 0, 0);
    o0 = wmma_f16(ap, load_B16(vcur, 32, 0, 0), o0);
    o1 = wmma_f16(ap, load_B16(vcur, 32, 0, 16), o1);
    o2 = wmma_f16(ap, load_B16(vcur, 32, 0, 32), o2);
    o3 = wmma_f16(ap, load_B16(vcur, 32, 0, 48), o3);

    stage_wait();   // next K/V tiles ready; releases double buffer (WAR)
  }

  long long cls = y[b];
  if (cls < 0) cls = 0;
  if (cls > 999) cls = 999;
  const float* mrow = mask + ((size_t)cls * 12 + h) * 64;
  const float m0v = mrow[(lane & 15)];
  const float m1v = mrow[16 + (lane & 15)];
  const float m2v = mrow[32 + (lane & 15)];
  const float m3v = mrow[48 + (lane & 15)];
#pragma unroll
  for (int r = 0; r < 8; ++r) {
    const float inv = 1.0f / li[r];
    const int row = (b << 10) + q0 + r + 8 * side;
    _Float16* orow = oh + (size_t)row * 768 + h * 64 + (lane & 15);
    orow[0]  = (_Float16)(o0[r] * inv * m0v);
    orow[16] = (_Float16)(o1[r] * inv * m1v);
    orow[32] = (_Float16)(o2[r] * inv * m2v);
    orow[48] = (_Float16)(o3[r] * inv * m3v);
  }
}

// ---------------------------------------------------------------------------
// K3: output projection  [8192,768] x [768,768]^T + bias -> f32 out
// ---------------------------------------------------------------------------
__global__ __launch_bounds__(256) void proj_gemm(
    const _Float16* __restrict__ ah,   // [8192, 768]
    const _Float16* __restrict__ wh,   // [768, 768]
    const float* __restrict__ bias,    // [768]
    float* __restrict__ out) {         // [8192, 768] f32
  __shared__ __align__(64) _Float16 bsh[2][64 * 32];
  const int lane = threadIdx.x & 31;
  const int wslot = threadIdx.x >> 5;
  const int NB = 768 / 64;                       // 12 col blocks
  const int m0 = (blockIdx.x / NB) * 128 + wslot * 16;
  const int n0 = (blockIdx.x % NB) * 64;

  v8f acc[4] = {{}, {}, {}, {}};
  gemm_core(ah, 768, wh, 768, m0, n0, 24, &bsh[0][0], &bsh[1][0], acc);

  const int side = lane >> 4;
#pragma unroll
  for (int j = 0; j < 4; ++j) {
    const int col = n0 + j * 16 + (lane & 15);
    const float bv = bias[col];
#pragma unroll
    for (int r = 0; r < 8; ++r)
      out[(size_t)(m0 + r + 8 * side) * 768 + col] = acc[j][r] + bv;
  }
}

// ---------------------------------------------------------------------------
// Host launcher
// ---------------------------------------------------------------------------
extern "C" void kernel_launch(void* const* d_in, const int* in_sizes, int n_in,
                              void* d_out, int out_size, void* d_ws, size_t ws_size,
                              hipStream_t stream) {
  (void)in_sizes; (void)n_in; (void)out_size; (void)ws_size;
  const float*      x      = (const float*)d_in[0];
  const long long*  y      = (const long long*)d_in[1];
  const float*      qkv_w  = (const float*)d_in[2];
  const float*      qkv_b  = (const float*)d_in[3];
  const float*      proj_w = (const float*)d_in[4];
  const float*      proj_b = (const float*)d_in[5];
  const float*      mask   = (const float*)d_in[6];
  float*            out    = (float*)d_out;

  char* ws = (char*)d_ws;
  _Float16* xh    = (_Float16*)(ws);              //  8192*768  = 12,582,912 B
  _Float16* wqkv  = (_Float16*)(ws + 12582912);   //  2304*768  =  3,538,944 B
  _Float16* wproj = (_Float16*)(ws + 16121856);   //   768*768  =  1,179,648 B
  _Float16* qh    = (_Float16*)(ws + 17301504);   //  [8,12,1024,64]
  _Float16* kh    = (_Float16*)(ws + 29884416);
  _Float16* vT    = (_Float16*)(ws + 42467328);   //  [8,12,64,1024]
  _Float16* ah    = (_Float16*)(ws + 55050240);   //  [8192,768]  (ends 67,633,152)

  cvt_f32_f16<<<(6291456 / 4 + 255) / 256, 256, 0, stream>>>(x, xh, 6291456 / 4);
  cvt_f32_f16<<<(1769472 / 4 + 255) / 256, 256, 0, stream>>>(qkv_w, wqkv, 1769472 / 4);
  cvt_f32_f16<<<(589824 / 4 + 255) / 256, 256, 0, stream>>>(proj_w, wproj, 589824 / 4);

  // 64 row-blocks x 36 col-blocks, 128x64 per block
  qkv_gemm<<<64 * 36, 256, 0, stream>>>(xh, wqkv, qkv_b, qh, kh, vT);
  // one (b,h) per 8 blocks; 8 q-blocks (waves) per block
  attn_fwd<<<768, 256, 0, stream>>>(qh, kh, vT, mask, y, ah);
  // 64 row-blocks x 12 col-blocks
  proj_gemm<<<64 * 12, 256, 0, stream>>>(ah, wproj, proj_b, out);
}
